// Attention_31499290149121
// MI455X (gfx1250) — matrix operand order repaired and checked by reading, not compiled, not run on hardware
//
#include <hip/hip_runtime.h>

// ---------------- types ----------------
typedef __bf16 bf16;
typedef __attribute__((ext_vector_type(16))) __bf16 bf16x16;
typedef __attribute__((ext_vector_type(8)))  __bf16 bf16x8;
typedef __attribute__((ext_vector_type(4)))  __bf16 bf16x4;
typedef __attribute__((ext_vector_type(8)))  float  f32x8;
typedef __attribute__((ext_vector_type(4)))  float  f32x4;

#define SEQ   4096
#define NH    12
#define HD    64
#define NF    768
#define OQKV  2304   // 3*NH*HD

// ---------------- helpers ----------------
__device__ __forceinline__ f32x8 zerof() {
    f32x8 z;
#pragma unroll
    for (int i = 0; i < 8; ++i) z[i] = 0.0f;
    return z;
}

__device__ __forceinline__ bf16x16 make_frag(bf16x8 lo, bf16x8 hi) {
    bf16x16 r;
#pragma unroll
    for (int i = 0; i < 8; ++i) { r[i] = lo[i]; r[i + 8] = hi[i]; }
    return r;
}

// A fragment (16x32 bf16), row-major source, leading dim ld (elements).
// ISA layout: lane(0-15): M=lane, K runs {half*8+0..7} and {16+half*8+0..7}.
__device__ __forceinline__ bf16x16 load_a(const bf16* M, int ld, int rowBase, int kBase, int lane) {
    int m = lane & 15, half = lane >> 4;
    const bf16* p = M + (size_t)(rowBase + m) * ld + kBase + half * 8;
    bf16x8 lo = *(const bf16x8*)p;
    bf16x8 hi = *(const bf16x8*)(p + 16);
    return make_frag(lo, hi);
}

// B fragment (32x16 bf16): B(k,n) = W[(colBase+n)*ld + kBase + k] (contiguous in k).
// ISA layout: lane: N=lane&15, K runs {half*8+0..7} and {16+half*8+0..7}.
__device__ __forceinline__ bf16x16 load_b(const bf16* W, int ld, int colBase, int kBase, int lane) {
    int n = lane & 15, half = lane >> 4;
    const bf16* p = W + (size_t)(colBase + n) * ld + kBase + half * 8;
    bf16x8 lo = *(const bf16x8*)p;
    bf16x8 hi = *(const bf16x8*)(p + 16);
    return make_frag(lo, hi);
}

__device__ __forceinline__ f32x8 wmma_bf16(bf16x16 a, bf16x16 b, f32x8 c) {
    return __builtin_amdgcn_wmma_f32_16x16x32_bf16(false, a, false, b, (short)0, c, false, false);
}

// CDNA5 async global->LDS copy (ASYNCcnt-tracked, no VGPR data path).
// vdst VGPR = LDS byte offset; vaddr pair = 64-bit global address; saddr = off (GV mode).
__device__ __forceinline__ void async_b128(unsigned lds_off, const bf16* g) {
    asm volatile("global_load_async_to_lds_b128 %0, %1, off"
                 :: "v"(lds_off), "v"(g)
                 : "memory");
}

// ---------------- kernel 1: f32 -> bf16 (x4 vectorized) ----------------
__global__ void cvt_bf16_kernel(const float* __restrict__ src, bf16* __restrict__ dst, int n4) {
    int i = blockIdx.x * 256 + threadIdx.x;
    if (i < n4) {
        f32x4 v = ((const f32x4*)src)[i];
        bf16x4 o;
#pragma unroll
        for (int j = 0; j < 4; ++j) o[j] = (bf16)v[j];
        ((bf16x4*)dst)[i] = o;
    }
}

// ---------------- kernel 2: fused QKV projection ----------------
// Each wave computes a 16x64 strip (one A fragment reused across 4 B tiles).
// o -> (h,d,s) with o = (h*64+d)*3+s.  Q pre-scaled by HD^-0.5, [h][n][64];
// K [h][n][64]; V stored TRANSPOSED [h][64][n] (makes P*V B-operands contiguous).
__global__ __launch_bounds__(256) void qkv_gemm_kernel(const bf16* __restrict__ xb,
                                                       const bf16* __restrict__ wb,
                                                       bf16* __restrict__ qb,
                                                       bf16* __restrict__ kb,
                                                       bf16* __restrict__ vtb) {
    int wave = (int)((blockIdx.x * blockDim.x + threadIdx.x) >> 5);
    int lane = threadIdx.x & 31;
    const int OG = OQKV / 64;           // 36 o-groups of 4 tiles
    int og = wave % OG;
    int nt = wave / OG;                 // 0..255

    f32x8 c0 = zerof(), c1 = zerof(), c2 = zerof(), c3 = zerof();
#pragma unroll 2
    for (int k = 0; k < NF; k += 32) {
        bf16x16 a = load_a(xb, NF, nt * 16, k, lane);
        c0 = wmma_bf16(a, load_b(wb, NF, og * 64 + 0,  k, lane), c0);
        c1 = wmma_bf16(a, load_b(wb, NF, og * 64 + 16, k, lane), c1);
        c2 = wmma_bf16(a, load_b(wb, NF, og * 64 + 32, k, lane), c2);
        c3 = wmma_bf16(a, load_b(wb, NF, og * 64 + 48, k, lane), c3);
    }

    int half = lane >> 4, N = lane & 15;
    const float scale = 0.125f;         // 64^-0.5
    f32x8 cc[4] = { c0, c1, c2, c3 };
#pragma unroll
    for (int tt = 0; tt < 4; ++tt) {
        int o = og * 64 + tt * 16 + N;
        int s = o % 3, d = (o / 3) % HD, h = o / (3 * HD);
#pragma unroll
        for (int r = 0; r < 8; ++r) {
            int n = nt * 16 + r + 8 * half;
            float v = cc[tt][r];
            if (s == 0)      qb[(size_t)h * SEQ * HD + (size_t)n * HD + d] = (bf16)(v * scale);
            else if (s == 1) kb[(size_t)h * SEQ * HD + (size_t)n * HD + d] = (bf16)v;
            else             vtb[(size_t)h * HD * SEQ + (size_t)d * SEQ + n] = (bf16)v;
        }
    }
}

// ---------------- kernel 3: flash attention ----------------
// Workgroup = 8 waves = 128 query rows of ONE head; K/V tiles are shared:
// async-copied global->LDS (double buffered) and consumed via ds_load_b128.
__global__ __launch_bounds__(256) void flash_attn_kernel(const bf16* __restrict__ qb,
                                                         const bf16* __restrict__ kb,
                                                         const bf16* __restrict__ vtb,
                                                         bf16* __restrict__ ob) {
    __shared__ bf16 Klds[2][32 * 64];   // [buf][kv_local * 64 + feat]      8 KB
    __shared__ bf16 Vlds[2][64 * 32];   // [buf][feat * 32 + kv_local]      8 KB
    __shared__ bf16 plds[8][16 * 32];   // per-wave P restage tile          8 KB

    int t    = threadIdx.x;
    int wid  = t >> 5;
    int lane = t & 31;
    int wave = blockIdx.x * 8 + wid;
    int qt = wave % (SEQ / 16);         // q 16-row block (8 consecutive per WG)
    int h  = wave / (SEQ / 16);         // head (uniform across the WG)

    const bf16* Q  = qb  + (size_t)h * SEQ * HD;
    const bf16* K  = kb  + (size_t)h * SEQ * HD;
    const bf16* VT = vtb + (size_t)h * HD * SEQ;

    bf16x16 aq0 = load_a(Q, HD, qt * 16, 0, lane);
    bf16x16 aq1 = load_a(Q, HD, qt * 16, 32, lane);

    f32x8 acc0 = zerof(), acc1 = zerof(), acc2 = zerof(), acc3 = zerof();
    float m[8], l[8];
#pragma unroll
    for (int r = 0; r < 8; ++r) { m[r] = -__builtin_huge_valf(); l[r] = 0.0f; }

    int half = lane >> 4, N = lane & 15;
    bf16* myp = plds[wid];
    const float L2E = 1.44269504088896340736f;

    // prefetch lambda: 256 threads x (16B K chunk + 16B V chunk) per 32-kv block
    auto prefetch = [&](int j, int buf) {
        {   // K tile: 32 rows x 64 feats; chunk t: row = t>>3, sub = t&7
            const bf16* g = K + (size_t)(j + (t >> 3)) * HD + (t & 7) * 8;
            async_b128((unsigned)(size_t)(&Klds[buf][0] + t * 8), g);
        }
        {   // V tile: 64 feats x 32 kv; chunk t: f = t>>2, sub = t&3
            const bf16* g = VT + (size_t)(t >> 2) * SEQ + j + (t & 3) * 8;
            async_b128((unsigned)(size_t)(&Vlds[buf][0] + t * 8), g);
        }
    };

    const int NJ = SEQ / 32;            // 128 kv blocks
    prefetch(0, 0);

    for (int jb = 0; jb < NJ; ++jb) {
        int cur = jb & 1;
        if (jb + 1 < NJ) {
            prefetch((jb + 1) * 32, cur ^ 1);
            asm volatile("s_wait_asynccnt 0x2" ::: "memory");  // drain current buffer's copies
        } else {
            asm volatile("s_wait_asynccnt 0x0" ::: "memory");
        }
        __syncthreads();                 // all threads' copies for `cur` visible

        const bf16* kl = &Klds[cur][0];
        const bf16* vl = &Vlds[cur][0];

        // S = Q * K^T (two 16x16 tiles over 32 kv columns), B from LDS
        f32x8 s0 = zerof(), s1 = zerof();
        s0 = wmma_bf16(aq0, load_b(kl, HD, 0,  0,  lane), s0);
        s0 = wmma_bf16(aq1, load_b(kl, HD, 0,  32, lane), s0);
        s1 = wmma_bf16(aq0, load_b(kl, HD, 16, 0,  lane), s1);
        s1 = wmma_bf16(aq1, load_b(kl, HD, 16, 32, lane), s1);

        // online softmax: row max across 16 N-lanes of each wave32 half
        float rmax[8];
#pragma unroll
        for (int r = 0; r < 8; ++r) rmax[r] = fmaxf(s0[r], s1[r]);
#pragma unroll
        for (int mask = 1; mask < 16; mask <<= 1)
#pragma unroll
            for (int r = 0; r < 8; ++r)
                rmax[r] = fmaxf(rmax[r], __shfl_xor(rmax[r], mask, 32));

        float alpha[8], p0[8], p1[8], psum[8];
#pragma unroll
        for (int r = 0; r < 8; ++r) {
            float mn = fmaxf(m[r], rmax[r]);
            alpha[r] = exp2f((m[r] - mn) * L2E);
            p0[r]    = exp2f((s0[r] - mn) * L2E);
            p1[r]    = exp2f((s1[r] - mn) * L2E);
            m[r] = mn;
            psum[r] = p0[r] + p1[r];
        }
#pragma unroll
        for (int mask = 1; mask < 16; mask <<= 1)
#pragma unroll
            for (int r = 0; r < 8; ++r)
                psum[r] += __shfl_xor(psum[r], mask, 32);
#pragma unroll
        for (int r = 0; r < 8; ++r) {
            l[r] = l[r] * alpha[r] + psum[r];
            acc0[r] *= alpha[r]; acc1[r] *= alpha[r];
            acc2[r] *= alpha[r]; acc3[r] *= alpha[r];
        }

        // Restage P: D-layout -> A-layout via per-wave LDS tile
#pragma unroll
        for (int r = 0; r < 8; ++r) {
            int mrow = r + 8 * half;
            myp[mrow * 32 + N]      = (bf16)p0[r];
            myp[mrow * 32 + 16 + N] = (bf16)p1[r];
        }
        asm volatile("s_wait_dscnt 0" ::: "memory");   // same-wave cross-lane LDS hazard
        bf16x16 ap;
        {
            int mm = lane & 15;
            const bf16* pp = myp + mm * 32 + half * 8;
            bf16x8 lo = *(const bf16x8*)pp;
            bf16x8 hi = *(const bf16x8*)(pp + 16);
            ap = make_frag(lo, hi);
        }

        // O += P * V  (V tile already transposed in LDS -> contiguous B frags)
        acc0 = wmma_bf16(ap, load_b(vl, 32, 0,  0, lane), acc0);
        acc1 = wmma_bf16(ap, load_b(vl, 32, 16, 0, lane), acc1);
        acc2 = wmma_bf16(ap, load_b(vl, 32, 32, 0, lane), acc2);
        acc3 = wmma_bf16(ap, load_b(vl, 32, 48, 0, lane), acc3);

        __syncthreads();                 // reads of buf `cur` done before next overwrite
    }

    // epilogue: normalize, store attn output bf16 [n, h*64 + d]
#pragma unroll
    for (int r = 0; r < 8; ++r) {
        int n = qt * 16 + r + 8 * half;
        float inv = 1.0f / l[r];
        bf16* row = ob + (size_t)n * NF + h * HD;
        row[0  + N] = (bf16)(acc0[r] * inv);
        row[16 + N] = (bf16)(acc1[r] * inv);
        row[32 + N] = (bf16)(acc2[r] * inv);
        row[48 + N] = (bf16)(acc3[r] * inv);
    }
}

// ---------------- kernel 4: output projection ----------------
// out[n,f] = sum_c attn[n,c] * w_out[f,c]; one wave = 16x64 strip.
__global__ __launch_bounds__(256) void out_gemm_kernel(const bf16* __restrict__ ob,
                                                       const bf16* __restrict__ wb,
                                                       float* __restrict__ out) {
    int wave = (int)((blockIdx.x * blockDim.x + threadIdx.x) >> 5);
    int lane = threadIdx.x & 31;
    const int FG = NF / 64;             // 12 f-groups
    int fg = wave % FG;
    int nt = wave / FG;

    f32x8 c0 = zerof(), c1 = zerof(), c2 = zerof(), c3 = zerof();
#pragma unroll 2
    for (int k = 0; k < NF; k += 32) {
        bf16x16 a = load_a(ob, NF, nt * 16, k, lane);
        c0 = wmma_bf16(a, load_b(wb, NF, fg * 64 + 0,  k, lane), c0);
        c1 = wmma_bf16(a, load_b(wb, NF, fg * 64 + 16, k, lane), c1);
        c2 = wmma_bf16(a, load_b(wb, NF, fg * 64 + 32, k, lane), c2);
        c3 = wmma_bf16(a, load_b(wb, NF, fg * 64 + 48, k, lane), c3);
    }

    int half = lane >> 4, N = lane & 15;
    f32x8 cc[4] = { c0, c1, c2, c3 };
#pragma unroll
    for (int tt = 0; tt < 4; ++tt)
#pragma unroll
        for (int r = 0; r < 8; ++r) {
            int n = nt * 16 + r + 8 * half;
            out[(size_t)n * NF + fg * 64 + tt * 16 + N] = cc[tt][r];
        }
}

// ---------------- launch ----------------
extern "C" void kernel_launch(void* const* d_in, const int* in_sizes, int n_in,
                              void* d_out, int out_size, void* d_ws, size_t ws_size,
                              hipStream_t stream) {
    const float* x     = (const float*)d_in[0];   // [1,4096,768]
    const float* w_qkv = (const float*)d_in[1];   // [2304,768]
    const float* w_out = (const float*)d_in[2];   // [768,768]
    float* out = (float*)d_out;                   // [1,4096,768]

    char* ws = (char*)d_ws;
    size_t off = 0;
    bf16* xb    = (bf16*)(ws + off); off += (size_t)SEQ * NF * 2;
    bf16* wqkvb = (bf16*)(ws + off); off += (size_t)OQKV * NF * 2;
    bf16* woutb = (bf16*)(ws + off); off += (size_t)NF * NF * 2;
    bf16* qb    = (bf16*)(ws + off); off += (size_t)NH * SEQ * HD * 2;
    bf16* kb    = (bf16*)(ws + off); off += (size_t)NH * SEQ * HD * 2;
    bf16* vtb   = (bf16*)(ws + off); off += (size_t)NH * SEQ * HD * 2;
    bf16* ob    = (bf16*)(ws + off); off += (size_t)SEQ * NF * 2;

    // 1) convert inputs to bf16 (x4 vectorized; all sizes divisible by 4)
    {
        int n1 = SEQ * NF / 4, n2 = OQKV * NF / 4, n3 = NF * NF / 4;
        cvt_bf16_kernel<<<(n1 + 255) / 256, 256, 0, stream>>>(x, xb, n1);
        cvt_bf16_kernel<<<(n2 + 255) / 256, 256, 0, stream>>>(w_qkv, wqkvb, n2);
        cvt_bf16_kernel<<<(n3 + 255) / 256, 256, 0, stream>>>(w_out, woutb, n3);
    }
    // 2) QKV projection: 256 n-tiles x 36 o-groups waves
    {
        int waves = (SEQ / 16) * (OQKV / 64);     // 9216
        qkv_gemm_kernel<<<waves / 8, 256, 0, stream>>>(xb, wqkvb, qb, kb, vtb);
    }
    // 3) flash attention: 12 heads * 256 q-blocks waves (8 per WG, same head)
    {
        int waves = NH * (SEQ / 16);              // 3072
        flash_attn_kernel<<<waves / 8, 256, 0, stream>>>(qb, kb, vtb, ob);
    }
    // 4) output projection: 256 n-tiles x 12 f-groups waves
    {
        int waves = (SEQ / 16) * (NF / 64);       // 3072
        out_gemm_kernel<<<waves / 8, 256, 0, stream>>>(ob, woutb, out);
    }
}
